// VectorQuantizer_86517821211380
// MI455X (gfx1250) — matrix operand order repaired and checked by reading, not compiled, not run on hardware
//
#include <hip/hip_runtime.h>
#include <math.h>

typedef __attribute__((ext_vector_type(16))) _Float16 v16h;
typedef __attribute__((ext_vector_type(8)))  _Float16 v8h;
typedef __attribute__((ext_vector_type(8)))  float    v8f;

#define NTOK   16384      // 16*32*32 tokens
#define KCB    8192       // codebook entries
#define DIM    256        // channel dim
#define BHW    1024       // 32*32 spatial per batch
#define NB     16         // batch

// ---------------- workspace layout (bytes) ----------------
#define WS_ZH    0                         // NTOK*DIM f16  (8 MB)  z token-major
#define WS_EH    (WS_ZH + NTOK*DIM*2)      // KCB*DIM f16   (4 MB)  emb
#define WS_E2    (WS_EH + KCB*DIM*2)       // KCB f32       ||e||^2
#define WS_HIST  (WS_E2 + KCB*4)           // KCB u32       histogram
#define WS_PART  (WS_HIST + KCB*4)         // 256 f32       loss partials
#define WS_IDX   (WS_PART + 256*4)         // NTOK i32      argmin indices

// ---------------- output layout (elements) ----------------
#define OUT_ZQ   0
#define OUT_LOSS ((size_t)NB*DIM*BHW)      // 4194304
#define OUT_PERP (OUT_LOSS + 1)
#define OUT_ENC  (OUT_PERP + 1)
#define OUT_IDX  (OUT_ENC + (size_t)NTOK*KCB)

// Zero the 512 MB one-hot region. Base is only 8B aligned -> float2 stores.
__global__ __launch_bounds__(256) void k_zero_enc(float2* __restrict__ enc2) {
  size_t base = ((size_t)blockIdx.x * 256 + threadIdx.x) * 8;
  float2 z2 = make_float2(0.f, 0.f);
#pragma unroll
  for (int i = 0; i < 8; ++i) enc2[base + i] = z2;
}

__global__ __launch_bounds__(256) void k_zero_hist(unsigned int* __restrict__ hist) {
  hist[blockIdx.x * 256 + threadIdx.x] = 0u;
}

// emb f32 -> f16, plus exact f32 row norms. One wave per codebook row.
__global__ __launch_bounds__(256) void k_conv_emb(const float* __restrict__ emb,
                                                  _Float16* __restrict__ eh,
                                                  float* __restrict__ e2) {
  const int lane = threadIdx.x & 31;
  const int wave = threadIdx.x >> 5;
  const int row  = blockIdx.x * 8 + wave;
  const float* src = emb + (size_t)row * DIM + lane * 8;
  float4 a = ((const float4*)src)[0];
  float4 b = ((const float4*)src)[1];
  v8h h;
  h[0] = (_Float16)a.x; h[1] = (_Float16)a.y; h[2] = (_Float16)a.z; h[3] = (_Float16)a.w;
  h[4] = (_Float16)b.x; h[5] = (_Float16)b.y; h[6] = (_Float16)b.z; h[7] = (_Float16)b.w;
  *(v8h*)(eh + (size_t)row * DIM + lane * 8) = h;
  float s = a.x*a.x + a.y*a.y + a.z*a.z + a.w*a.w
          + b.x*b.x + b.y*b.y + b.z*b.z + b.w*b.w;
#pragma unroll
  for (int off = 16; off >= 1; off >>= 1) s += __shfl_xor(s, off, 32);
  if (lane == 0) e2[row] = s;
}

// z [b,c,h,w] f32 -> zh [n=(b,h,w)][c] f16 (coalesced reads).
__global__ __launch_bounds__(256) void k_conv_z(const float* __restrict__ z,
                                                _Float16* __restrict__ zh) {
  const int e  = blockIdx.x * 256 + threadIdx.x;
  const int b  = e >> 18;              // 256*1024 elems per batch
  const int c  = (e >> 10) & (DIM - 1);
  const int hw = e & (BHW - 1);
  const int n  = b * BHW + hw;
  zh[(size_t)n * DIM + c] = (_Float16)z[e];
}

// Core: WMMA distance GEMM + per-row argmin over the codebook.
// 4 waves/block, 16 z-rows per wave. Codebook streamed through a
// double-buffered LDS tile with GLOBAL_LOAD_ASYNC_TO_LDS_B128 (ASYNCcnt).
#define BROWS   64
#define BSTRIDE (DIM + 8)             // pad 16B/row -> conflict-free ds_read_b128
#define BUFH    (BROWS * BSTRIDE)     // halves per buffer
__global__ __launch_bounds__(128) void k_vq_argmin(
    const _Float16* __restrict__ zh, const _Float16* __restrict__ eh,
    const float* __restrict__ e2, int* __restrict__ widx,
    float* __restrict__ oidxf, unsigned int* __restrict__ hist) {
  __shared__ __align__(16) _Float16 Bs[2 * BUFH];   // ~66 KB double buffer
  const int tid      = threadIdx.x;
  const int lane     = tid & 31;
  const int wave     = tid >> 5;
  const int half_sel = lane >> 4;        // 0: lanes 0-15, 1: lanes 16-31
  const int m        = lane & 15;
  const int col_l    = lane & 15;
  const int koff     = half_sel * 8;
  const int rowbase  = (blockIdx.x * 4 + wave) * 16;

  // Per-thread async-copy addressing: thread covers rows (tid>>5)+4i, 16B col (tid&31).
  const unsigned lds_thr = (unsigned)(uintptr_t)Bs
                         + (unsigned)((tid >> 5) * (BSTRIDE * 2) + (tid & 31) * 16);
  const _Float16* gthr = eh + (size_t)(tid >> 5) * DIM + (size_t)(tid & 31) * 8;

#define ISSUE_TILE(MT, BUF)                                                     \
  do {                                                                          \
    _Pragma("unroll")                                                           \
    for (int i_ = 0; i_ < 16; ++i_) {                                           \
      unsigned loff_ = lds_thr + (unsigned)(BUF) * (BUFH * 2)                   \
                     + (unsigned)(i_ * 4 * BSTRIDE * 2);                        \
      unsigned long long ga_ = (unsigned long long)(uintptr_t)                  \
          (gthr + (size_t)(MT) * BROWS * DIM + (size_t)i_ * 4 * DIM);           \
      asm volatile("global_load_async_to_lds_b128 %0, %1, off"                  \
                   :: "v"(loff_), "v"(ga_) : "memory");                         \
    }                                                                           \
  } while (0)

  // A tile (16x256 f16) in registers, CDNA5 16-bit A-matrix layout:
  // lane<16: row m, K in {32c+0..7, 32c+16..23}; lane>=16: +8 offset.
  v16h A[8];
  {
    const _Float16* arow = zh + (size_t)(rowbase + m) * DIM;
#pragma unroll
    for (int c = 0; c < 8; ++c) {
      const int kb = c * 32 + half_sel * 8;
      ((float4*)&A[c])[0] = *(const float4*)(arow + kb);
      ((float4*)&A[c])[1] = *(const float4*)(arow + kb + 16);
    }
  }

  float minval[8]; int minidx[8];
#pragma unroll
  for (int i = 0; i < 8; ++i) { minval[i] = 3.4e38f; minidx[i] = 0; }

  ISSUE_TILE(0, 0);                             // prologue: tile 0 -> buf 0

  for (int mt = 0; mt < KCB / BROWS; ++mt) {    // 128 tiles of 64 codebook rows
    if (mt + 1 < KCB / BROWS) {
      ISSUE_TILE(mt + 1, (mt + 1) & 1);         // prefetch next tile async
      asm volatile("s_wait_asynccnt 0x10" ::: "memory");  // current tile landed
    } else {
      asm volatile("s_wait_asynccnt 0x0" ::: "memory");
    }
    __syncthreads();

    // Hoist ||e||^2 for this tile's 64 columns (4 per lane).
    float ce2[4];
#pragma unroll
    for (int s = 0; s < 4; ++s) ce2[s] = e2[mt * BROWS + s * 16 + col_l];

    const _Float16* bbuf = Bs + (mt & 1) * BUFH;
#pragma unroll
    for (int s = 0; s < 4; ++s) {               // 4 sub-tiles of 16 columns
      // B layout: lane<16 holds col=col_l, k in {0..7,16..23} per chunk; lane>=16: +8 offset.
      const _Float16* brow = bbuf + (s * 16 + col_l) * BSTRIDE + koff;
      v16h B[8];
#pragma unroll
      for (int c = 0; c < 8; ++c) {             // batch all 16 ds_read_b128
        ((float4*)&B[c])[0] = *(const float4*)(brow + c * 32);
        ((float4*)&B[c])[1] = *(const float4*)(brow + c * 32 + 16);
      }
      v8f acc = {};
#pragma unroll
      for (int c = 0; c < 8; ++c)               // 8 back-to-back WMMAs
        acc = __builtin_amdgcn_wmma_f32_16x16x32_f16(
            false, A[c], false, B[c], (short)0, acc, false, false);

      const int col = mt * BROWS + s * 16 + col_l;
#pragma unroll
      for (int i = 0; i < 8; ++i) {
        float d = __builtin_fmaf(-2.0f, acc[i], ce2[s]);   // ||e||^2 - 2 z.e
        if (d < minval[i]) { minval[i] = d; minidx[i] = col; }
      }
    }
    __syncthreads();
  }

  // Cross-lane reduce over the 16 lanes sharing each row; ties -> lowest index.
#pragma unroll
  for (int i = 0; i < 8; ++i) {
    float v = minval[i]; int id = minidx[i];
#pragma unroll
    for (int off = 8; off >= 1; off >>= 1) {
      float ov = __shfl_xor(v, off, 32);
      int   oi = __shfl_xor(id, off, 32);
      if (ov < v || (ov == v && oi < id)) { v = ov; id = oi; }
    }
    if (col_l == 0) {
      const int row = rowbase + i + 8 * half_sel;
      widx[row]  = id;
      oidxf[row] = (float)id;
      atomicAdd(&hist[id], 1u);
    }
  }
#undef ISSUE_TILE
}

// Gather z_q, scatter one-hot 1.0s, per-block deterministic loss partials.
__global__ __launch_bounds__(256) void k_gather_loss(
    const float* __restrict__ z, const float* __restrict__ emb,
    const int* __restrict__ idx, float* __restrict__ zq,
    float* __restrict__ enc, float* __restrict__ part) {
  __shared__ float red[256];
  const int tid = threadIdx.x;
  const int b   = blockIdx.x >> 4;
  const int hw  = (blockIdx.x & 15) * 64 + (tid & 63);
  const int cg  = tid >> 6;                  // 4 channel groups
  const int n   = b * BHW + hw;
  const int k   = idx[n];
  if (cg == 0) enc[(size_t)n * KCB + k] = 1.0f;
  const float* erow = emb + (size_t)k * DIM;
  float ls = 0.f;
  for (int c = cg; c < DIM; c += 4) {
    const size_t off = (((size_t)(b * DIM + c)) << 10) + hw;
    const float q = erow[c];
    const float d = q - z[off];
    zq[off] = q;                             // straight-through value == z_q
    ls += d * d;
  }
  red[tid] = ls;
  __syncthreads();
  for (int st = 128; st > 0; st >>= 1) {
    if (tid < st) red[tid] += red[tid + st];
    __syncthreads();
  }
  if (tid == 0) part[blockIdx.x] = red[0];
}

__global__ __launch_bounds__(256) void k_finalize(
    const unsigned int* __restrict__ hist, const float* __restrict__ part,
    float* __restrict__ out_loss, float* __restrict__ out_perp) {
  __shared__ float red[256];
  const int tid = threadIdx.x;
  float s = 0.f;
  for (int i = tid; i < KCB; i += 256) {
    const float p = (float)hist[i] * (1.0f / (float)NTOK);
    s += p * logf(p + 1e-10f);
  }
  red[tid] = s; __syncthreads();
  for (int st = 128; st > 0; st >>= 1) {
    if (tid < st) red[tid] += red[tid + st];
    __syncthreads();
  }
  const float ent = red[0];
  __syncthreads();
  red[tid] = part[tid]; __syncthreads();
  for (int st = 128; st > 0; st >>= 1) {
    if (tid < st) red[tid] += red[tid + st];
    __syncthreads();
  }
  if (tid == 0) {
    *out_perp = expf(-ent);
    *out_loss = 1.25f * red[0] / (float)((size_t)NB * DIM * BHW);  // (1+BETA)*MSE
  }
}

extern "C" void kernel_launch(void* const* d_in, const int* in_sizes, int n_in,
                              void* d_out, int out_size, void* d_ws, size_t ws_size,
                              hipStream_t stream) {
  const float* z   = (const float*)d_in[0];   // [16,256,32,32]
  const float* emb = (const float*)d_in[1];   // [8192,256]
  float* outf = (float*)d_out;
  char*  ws   = (char*)d_ws;

  _Float16*     zh    = (_Float16*)(ws + WS_ZH);
  _Float16*     eh    = (_Float16*)(ws + WS_EH);
  float*        e2    = (float*)(ws + WS_E2);
  unsigned int* hist  = (unsigned int*)(ws + WS_HIST);
  float*        part  = (float*)(ws + WS_PART);
  int*          widx  = (int*)(ws + WS_IDX);

  float* zq    = outf + OUT_ZQ;
  float* enc   = outf + OUT_ENC;
  float* oidxf = outf + OUT_IDX;

  // 512MB one-hot zero: 32768 blocks * 256 thr * 8 float2 * 2 = 134217728 floats
  k_zero_enc   <<<32768,            256, 0, stream>>>((float2*)enc);
  k_zero_hist  <<<KCB / 256,        256, 0, stream>>>(hist);
  k_conv_emb   <<<KCB / 8,          256, 0, stream>>>(emb, eh, e2);
  k_conv_z     <<<(NB*DIM*BHW)/256, 256, 0, stream>>>(z, zh);
  k_vq_argmin  <<<NTOK / 64,        128, 0, stream>>>(zh, eh, e2, widx, oidxf, hist);
  k_gather_loss<<<NB * 16,          256, 0, stream>>>(z, emb, widx, zq, enc, part);
  k_finalize   <<<1,                256, 0, stream>>>(hist, part,
                                                      outf + OUT_LOSS, outf + OUT_PERP);
}
